// WindowAttention_16166256902425
// MI455X (gfx1250) — matrix operand order repaired and checked by reading, not compiled, not run on hardware
//
#include <hip/hip_runtime.h>
#include <hip/hip_bf16.h>
#include <stddef.h>

// ---------------------------------------------------------------------------
// Swin shifted-window attention for MI455X (gfx1250, wave32, WMMA).
//   x:(32,56,56,384) f32 -> out:(32,56,56,384) f32
//   windows: 7x7, 2048 windows, 12 heads, hd=32 (== WMMA K depth for bf16)
// ---------------------------------------------------------------------------

typedef __attribute__((ext_vector_type(16))) __bf16 v16bf;
typedef __attribute__((ext_vector_type(8)))  __bf16 v8bf;
typedef __attribute__((ext_vector_type(8)))  float  v8f;

#define CDIM   384
#define QKVN   1152
#define HEADS  12
#define HD     32
#define WW     49
#define NWIN   2048
#define QSTR   1160   // qkv LDS row stride (halves); 2320B, 16B aligned, bank-skewed
#define XSTR   392    // x LDS row stride (halves); 784B
#define SSTR   68     // score row stride (f32); 272B

#define USE_ASYNC_LDS 1

// CDNA5 async global->LDS copy (ASYNCcnt-tracked). VDST is the LDS byte
// offset: on AMDGPU the low 32 bits of a generic pointer to LDS are the AS3
// offset. VADDR is the 64-bit global address; "off" = no SADDR.
__device__ __forceinline__ void copy16_g2l(const __bf16* src, __bf16* dst) {
#if USE_ASYNC_LDS
  unsigned lds = (unsigned)(size_t)dst;
  unsigned long long ga = (unsigned long long)(size_t)src;
  asm volatile("global_load_async_to_lds_b128 %0, %1, off"
               :: "v"(lds), "v"(ga) : "memory");
#else
  *(v8bf*)dst = *(const v8bf*)src;
#endif
}

__device__ __forceinline__ void async_copy_fence() {
#if USE_ASYNC_LDS
  asm volatile("s_wait_asynccnt 0x0" ::: "memory");
#endif
}

__device__ __forceinline__ v8f wmma_bf16(v16bf a, v16bf b, v8f c) {
  return __builtin_amdgcn_wmma_f32_16x16x32_bf16(false, a, false, b,
                                                 (short)0, c, false, false);
}

// A-matrix 16x32 bf16 fragment (contiguous K, matches B): lane<16 -> K 0..15,
// lane>=16 -> K 16..31; K-contiguous -> lowers to 2x ds_load_b128.
__device__ __forceinline__ v16bf load_afrag(const __bf16* buf, int stride,
                                            int m0, int k0, int lane) {
  const int half = lane >> 4;
  const __bf16* row = buf + (size_t)(m0 + (lane & 15)) * stride + k0;
  v16bf a;
#pragma unroll
  for (int j = 0; j < 8; ++j) {
    int k = half * 16 + 2 * j;
    a[2 * j]     = row[k];
    a[2 * j + 1] = row[k + 1];
  }
  return a;
}

// B-matrix 32x16 bf16 fragment, K-contiguous source: lane holds col n=lane%16,
// K = (lane/16)*16 + e. Caller pre-offsets base to (n, (lane/16)*16).
__device__ __forceinline__ v16bf load_bfrag_contig(const __bf16* base) {
  v8bf lo = *(const v8bf*)(base);
  v8bf hi = *(const v8bf*)(base + 8);
  v16bf b;
#pragma unroll
  for (int i = 0; i < 8; ++i) { b[i] = lo[i]; b[8 + i] = hi[i]; }
  return b;
}

// B-matrix fragment with strided K (used for V which is stored token-major).
__device__ __forceinline__ v16bf load_bfrag_strided(const __bf16* base, int stride) {
  v16bf b;
#pragma unroll
  for (int e = 0; e < 16; ++e) b[e] = base[(size_t)e * stride];
  return b;
}

// ---------------------------------------------------------------------------
// Kernel 0: weights fp32 -> bf16 (then they live in the 192MB L2)
// ---------------------------------------------------------------------------
__global__ __launch_bounds__(256) void convert_weights(
    const float* __restrict__ wqkv, const float* __restrict__ wproj,
    __bf16* __restrict__ owqkv, __bf16* __restrict__ owproj) {
  int i = blockIdx.x * 256 + threadIdx.x;
  if (i < QKVN * CDIM) owqkv[i] = (__bf16)wqkv[i];
  if (i < CDIM * CDIM) owproj[i] = (__bf16)wproj[i];
}

// ---------------------------------------------------------------------------
// Kernel 1: fused roll + window + QKV GEMM + attention.  1 block = 1 window.
// Dynamic LDS: [ qkv 64xQSTR bf16 | union( x 64xXSTR bf16, S 8x64xSSTR f32 ) ]
// ---------------------------------------------------------------------------
extern __shared__ char smem[];

__global__ __launch_bounds__(256) void fused_win_attn(
    const float* __restrict__ x, const __bf16* __restrict__ wqkv,
    const float* __restrict__ bqkv, const float* __restrict__ rel,
    __bf16* __restrict__ Obuf) {
  __bf16* qkv = (__bf16*)smem;                       // 64 x QSTR halves
  char*  ph2  = smem + (size_t)64 * QSTR * 2;        // union region
  __bf16* xb  = (__bf16*)ph2;                        // 64 x XSTR halves
  float*  sb  = (float*)ph2;                         // 8 x 64 x SSTR f32

  const int win  = blockIdx.x;          // 0..2047
  const int b    = win >> 6;
  const int wh   = (win >> 3) & 7;
  const int wc   = win & 7;
  const int tid  = threadIdx.x;
  const int wv   = tid >> 5;            // wave id (wave32)
  const int lane = tid & 31;

  // ---- phase 1: gather rolled window into LDS as bf16, zero-padded to 64 rows
  for (int i = tid; i < 64 * XSTR; i += 256) xb[i] = (__bf16)0.0f;
  __syncthreads();
  for (int i = tid; i < WW * CDIM; i += 256) {
    int t = i / CDIM, c = i - t * CDIM;
    int ty = t / 7, tx = t - ty * 7;
    int h = (wh * 7 + ty + 3) % 56;     // roll(-3)
    int w = (wc * 7 + tx + 3) % 56;
    xb[t * XSTR + c] = (__bf16)x[(((size_t)b * 56 + h) * 56 + w) * CDIM + c];
  }
  __syncthreads();

  // ---- phase 2: QKV(64x1152) = X(64x384) @ Wqkv^T + b, kept in LDS as bf16.
  // Wave wv owns N-tiles [wv*9, wv*9+9).  B fragments are register-cached and
  // each weight element is read from L2 exactly once per block; A fragments
  // stream from LDS (2x ds_load_b128 each).  nt kept rolled to bound pressure.
#pragma unroll 1
  for (int nt = 0; nt < 9; ++nt) {
    const int n0 = (wv * 9 + nt) * 16;
    const int n  = n0 + (lane & 15);
    const float bias = bqkv[n];
    v16bf bfr[12];
#pragma unroll
    for (int ks = 0; ks < 12; ++ks)
      bfr[ks] = load_bfrag_contig(wqkv + (size_t)n * CDIM + ks * 32 +
                                  (lane >> 4) * 16);
#pragma unroll 2
    for (int mt = 0; mt < 4; ++mt) {
      v8f acc = {};
#pragma unroll
      for (int ks = 0; ks < 12; ++ks)
        acc = wmma_bf16(load_afrag(xb, XSTR, mt * 16, ks * 32, lane),
                        bfr[ks], acc);
#pragma unroll
      for (int r = 0; r < 8; ++r) {
        int m = mt * 16 + (lane >> 4) * 8 + r;
        qkv[(size_t)m * QSTR + n] = (__bf16)(acc[r] + bias);
      }
    }
  }
  __syncthreads();

  // ---- phase 3: attention. Wave wv handles heads {wv, wv+8}.
  const float scale = 0.17677669529663687f;   // 1/sqrt(32)
  float*  mysb = sb + (size_t)wv * 64 * SSTR;
  __bf16* myp  = (__bf16*)mysb;               // bf16 P overlaid on S rows

#pragma unroll 1
  for (int h = wv; h < HEADS; h += 8) {
    // S = Q K^T : hd==32 -> single WMMA per 16x16 tile.
#pragma unroll 1
    for (int mt = 0; mt < 4; ++mt) {
      v16bf qa = load_afrag(qkv, QSTR, mt * 16, h * HD, lane);
#pragma unroll
      for (int nt = 0; nt < 4; ++nt) {
        const int kt = nt * 16 + (lane & 15);
        const __bf16* kb = qkv + (size_t)kt * QSTR + CDIM + h * HD +
                           (lane >> 4) * 16;
        v8f s = {};
        s = wmma_bf16(qa, load_bfrag_contig(kb), s);
        // fused epilogue: scale + rel-pos bias + shift masks + padding
        int ky = 0, kx = 0;
        if (kt < WW) { ky = kt / 7; kx = kt - ky * 7; }
#pragma unroll
        for (int r = 0; r < 8; ++r) {
          int qt = mt * 16 + (lane >> 4) * 8 + r;
          float v;
          if (qt >= WW)      v = (kt == 0) ? 0.0f : -1e30f;  // pad rows: delta
          else if (kt >= WW) v = -1e30f;                      // pad cols: -inf
          else {
            int qy = qt / 7, qx = qt - qy * 7;
            int idx = (qy - ky + 6) * 13 + (qx - kx + 6);
            v = s[r] * scale + rel[idx * HEADS + h];
            if (wh == 7 && ((qt >= 28) != (kt >= 28))) v = -1e30f;  // ul mask
            if (wc == 7 && ((qx >= 4)  != (kx >= 4)))  v = -1e30f;  // lr mask
          }
          mysb[qt * SSTR + kt] = v;
        }
      }
    }
    // softmax per row; write P as bf16 in place (2c write never passes 4c read)
    for (int row = lane; row < 64; row += 32) {
      float* rp = mysb + row * SSTR;
      float mx = -1e30f;
      for (int c = 0; c < 64; ++c) mx = fmaxf(mx, rp[c]);
      float sum = 0.0f;
      for (int c = 0; c < 64; ++c) sum += __expf(rp[c] - mx);
      float inv = 1.0f / sum;
      __bf16* pr = (__bf16*)rp;
      for (int c = 0; c < 64; ++c) pr[c] = (__bf16)(__expf(rp[c] - mx) * inv);
    }
    // O_head = P(64x64) @ V(64x32)
#pragma unroll 1
    for (int mt = 0; mt < 4; ++mt) {
#pragma unroll
      for (int nt3 = 0; nt3 < 2; ++nt3) {
        v8f acc = {};
#pragma unroll
        for (int ks = 0; ks < 2; ++ks) {
          v16bf pa = load_afrag(myp, SSTR * 2, mt * 16, ks * 32, lane);
          const __bf16* vb = qkv + (size_t)(ks * 32 + (lane >> 4) * 16) * QSTR +
                             2 * CDIM + h * HD + nt3 * 16 + (lane & 15);
          acc = wmma_bf16(pa, load_bfrag_strided(vb, QSTR), acc);
        }
#pragma unroll
        for (int r = 0; r < 8; ++r) {
          int m = mt * 16 + (lane >> 4) * 8 + r;
          if (m < WW) {
            int col = h * HD + nt3 * 16 + (lane & 15);
            Obuf[((size_t)win * 64 + m) * CDIM + col] = (__bf16)acc[r];
          }
        }
      }
    }
  }
}

// ---------------------------------------------------------------------------
// Kernel 2: un-window + un-roll + output projection.
// Grid (1568, 3): block = 64 rows x 128 cols; each wave owns one 16-col tile.
// ---------------------------------------------------------------------------
__global__ __launch_bounds__(256) void proj_kernel(
    const __bf16* __restrict__ Obuf, const __bf16* __restrict__ wproj,
    const float* __restrict__ bproj, float* __restrict__ out) {
  __shared__ __bf16 Ab[64 * 40];      // 64 rows x 32 K (stride 40 -> 16B align)
  __shared__ __bf16 Bb[128 * 40];     // 128 n-rows x 32 K
  __shared__ int rowsrc[64];

  const int tid = threadIdx.x, wv = tid >> 5, lane = tid & 31;
  const int m0 = blockIdx.x * 64;
  const int nb = blockIdx.y * 128;    // N range of this block

  if (tid < 64) {
    int r = m0 + tid;
    int b = r / 3136, rem = r - b * 3136;
    int h = rem / 56, w = rem - h * 56;
    int hs = (h + 53) % 56, ws = (w + 53) % 56;      // inverse of roll(-3)
    int win = b * 64 + (hs / 7) * 8 + (ws / 7);
    int tok = (hs % 7) * 7 + (ws % 7);
    rowsrc[tid] = win * 64 + tok;
  }
  __syncthreads();

  v8f acc[4] = {};
#pragma unroll 1
  for (int ks = 0; ks < 12; ++ks) {
    __syncthreads();
    {  // stage A tile: 64 rows x 32 halves (gathered via rowsrc), async copy
      int row = tid >> 2, seg = tid & 3;
      const __bf16* src = Obuf + (size_t)rowsrc[row] * CDIM + ks * 32 + seg * 8;
      copy16_g2l(src, Ab + row * 40 + seg * 8);
      if (ks < 11) __builtin_prefetch(src + 32, 0, 3);   // global_prefetch_b8
    }
#pragma unroll
    for (int t = tid; t < 128 * 4; t += 256) {  // stage B tile (K-contig rows)
      int n = t >> 2, seg = t & 3;
      copy16_g2l(wproj + (size_t)(nb + n) * CDIM + ks * 32 + seg * 8,
                 Bb + n * 40 + seg * 8);
    }
    async_copy_fence();
    __syncthreads();
    const __bf16* bp = Bb + (size_t)(wv * 16 + (lane & 15)) * 40 +
                       (lane >> 4) * 16;
    v16bf bf_ = load_bfrag_contig(bp);
#pragma unroll
    for (int mt = 0; mt < 4; ++mt)
      acc[mt] = wmma_bf16(load_afrag(Ab, 40, mt * 16, 0, lane), bf_, acc[mt]);
  }
  const int n = nb + wv * 16 + (lane & 15);
  const float bias = bproj[n];
#pragma unroll
  for (int mt = 0; mt < 4; ++mt)
#pragma unroll
    for (int r = 0; r < 8; ++r) {
      int m = mt * 16 + (lane >> 4) * 8 + r;
      out[(size_t)(m0 + m) * CDIM + n] = acc[mt][r] + bias;
    }
}

// ---------------------------------------------------------------------------
extern "C" void kernel_launch(void* const* d_in, const int* in_sizes, int n_in,
                              void* d_out, int out_size, void* d_ws, size_t ws_size,
                              hipStream_t stream) {
  (void)in_sizes; (void)n_in; (void)out_size; (void)ws_size;
  const float* x     = (const float*)d_in[0];
  const float* wqkv  = (const float*)d_in[1];
  const float* bqkv  = (const float*)d_in[2];
  const float* wproj = (const float*)d_in[3];
  const float* bproj = (const float*)d_in[4];
  const float* rel   = (const float*)d_in[5];
  float* out = (float*)d_out;

  char* ws = (char*)d_ws;
  __bf16* wqkv_bf  = (__bf16*)ws;                                  // 884736 B
  __bf16* wproj_bf = (__bf16*)(ws + 884736);                       // 294912 B
  __bf16* Obuf     = (__bf16*)(ws + 884736 + 294912);              // ~100.7 MB

  convert_weights<<<1728, 256, 0, stream>>>(wqkv, wproj, wqkv_bf, wproj_bf);

  // 64*QSTR*2 (qkv) + 8*64*SSTR*4 (scores) = 148480 + 139264 = 287744 B
  // (fits WGP's 320KB LDS; one workgroup per WGP)
  size_t smem = (size_t)64 * QSTR * 2 + (size_t)8 * 64 * SSTR * 4;
  fused_win_attn<<<NWIN, 256, smem, stream>>>(x, wqkv_bf, bqkv, rel, Obuf);

  proj_kernel<<<dim3(100352 / 64, 3), 256, 0, stream>>>(Obuf, wproj_bf, bproj, out);
}
